// LlamaAttention_86036784873995
// MI455X (gfx1250) — compile-verified, matrix operand order checked by reading, not run on hardware
//
#include <hip/hip_runtime.h>
#include <hip/hip_bf16.h>
#include <cstdint>

// ---- problem constants ----
#define BATCH 2
#define SEQ   2048
#define HID   4096
#define NH    32
#define NKV   8
#define HD    128
#define QSZ   4096
#define KVSZ  1024
#define NQKV  6144   // QSZ + 2*KVSZ

typedef __attribute__((ext_vector_type(16))) __bf16 v16bf;
typedef __attribute__((ext_vector_type(8)))  float  v8f;

union Frag { uint4 q[2]; v16bf v; };   // 32B: one WMMA bf16 A/B operand

__device__ __forceinline__ unsigned short f2bf(float f) {
    unsigned int u = __float_as_uint(f);
    unsigned int r = u + 0x7FFFu + ((u >> 16) & 1u);   // round-to-nearest-even
    return (unsigned short)(r >> 16);
}

// CDNA5 async global->LDS copy (ASYNCcnt-tracked, completes in order).
// lds_off = addr[31:0] of the generic LDS pointer (aperture rule). 16B per lane.
__device__ __forceinline__ void async_load_b128(unsigned lds_off, const void* gaddr) {
    asm volatile("global_load_async_to_lds_b128 %0, %1, off"
                 :: "v"(lds_off), "v"(gaddr) : "memory");
}
// drain everything
__device__ __forceinline__ void wait_async0() {
    asm volatile("s_wait_asynccnt 0x0" ::: "memory");
}
// leave the most recent 4 async loads (next tile) in flight
__device__ __forceinline__ void wait_async4() {
    asm volatile("s_wait_asynccnt 0x4" ::: "memory");
}

// ---------------- cast f32 -> bf16, 4 elems/thread ----------------
__global__ __launch_bounds__(256) void castk(const float* __restrict__ src,
                                             unsigned short* __restrict__ dst, int n4) {
    int i = blockIdx.x * blockDim.x + threadIdx.x;
    if (i < n4) {
        float4 f = ((const float4*)src)[i];
        ushort4 o;
        o.x = f2bf(f.x); o.y = f2bf(f.y); o.z = f2bf(f.z); o.w = f2bf(f.w);
        ((ushort4*)dst)[i] = o;
    }
}

// ------------- cast + transpose: src f32 [K][N] -> dst bf16 [N][K] -------------
__global__ __launch_bounds__(256) void castT(const float* __restrict__ src,
                                             unsigned short* __restrict__ dst,
                                             int K, int N) {
    __shared__ unsigned short t[32][33];
    int n0 = blockIdx.x * 32, k0 = blockIdx.y * 32;
    int tx = threadIdx.x & 31, ty = threadIdx.x >> 5;   // 32 x 8
    #pragma unroll
    for (int i = 0; i < 32; i += 8)
        t[ty + i][tx] = f2bf(src[(size_t)(k0 + ty + i) * N + n0 + tx]);
    __syncthreads();
    #pragma unroll
    for (int i = 0; i < 32; i += 8)
        dst[(size_t)(n0 + ty + i) * K + k0 + tx] = t[tx][ty + i];
}

// ---------------- bf16 WMMA GEMM: C(MxN,f32) = A(MxK) * Bt(NxK)^T ----------------
// 256 threads = 8 waves (4M x 2N); block tile 128M x 128N; wave tile 32M x 64N;
// k-step 32. Double-buffered LDS staging with CDNA5 async global->LDS b128 copies:
// tile t+1's DMA flies underneath tile t's WMMAs (in-order ASYNCcnt completion
// lets s_wait_asynccnt 4 drain only the current tile).
__global__ __launch_bounds__(256) void gemm_bf16f32(
    const unsigned short* __restrict__ A,   // M x K, row-major bf16
    const unsigned short* __restrict__ Bt,  // N x K, row-major bf16 (pre-transposed)
    float* __restrict__ C, int M, int N, int K)
{
    __shared__ __align__(16) unsigned short As[2][128][40];   // [buf][m][k], padded rows
    __shared__ __align__(16) unsigned short Bs[2][128][40];   // [buf][n][k], padded rows

    const int tid  = threadIdx.x;
    const int lane = tid & 31;
    const int w    = tid >> 5;
    const int wm   = w & 3;          // 0..3 -> 32-row slice
    const int wn   = w >> 2;         // 0..1 -> 64-col slice
    const int bm   = blockIdx.y * 128;
    const int bn   = blockIdx.x * 128;

    // per-thread staging coordinates: 2 b128 chunks per tile per matrix
    const int r0 = (tid * 2) >> 2, c0 = ((tid * 2) & 3) * 8;
    const int r1 = (tid * 2 + 1) >> 2, c1 = ((tid * 2 + 1) & 3) * 8;
    const unsigned loff0 = (unsigned)(r0 * 40 + c0) * 2u;
    const unsigned loff1 = (unsigned)(r1 * 40 + c1) * 2u;
    const unsigned asl[2] = { (unsigned)(uintptr_t)&As[0][0][0],
                              (unsigned)(uintptr_t)&As[1][0][0] };
    const unsigned bsl[2] = { (unsigned)(uintptr_t)&Bs[0][0][0],
                              (unsigned)(uintptr_t)&Bs[1][0][0] };
    const unsigned short* Ar0 = A  + (size_t)(bm + r0) * K;
    const unsigned short* Ar1 = A  + (size_t)(bm + r1) * K;
    const unsigned short* Br0 = Bt + (size_t)(bn + r0) * K;
    const unsigned short* Br1 = Bt + (size_t)(bn + r1) * K;

    v8f acc[2][4] = {};
    const int nk = K >> 5;

    // prologue: stage tile 0 into buffer 0
    async_load_b128(asl[0] + loff0, Ar0 + c0);
    async_load_b128(asl[0] + loff1, Ar1 + c1);
    async_load_b128(bsl[0] + loff0, Br0 + c0);
    async_load_b128(bsl[0] + loff1, Br1 + c1);

    for (int t = 0; t < nk; ++t) {
        const int buf = t & 1;
        if (t + 1 < nk) {
            // stage next tile into the other buffer (safe: barrier at end of
            // previous iteration means everyone finished reading it)
            const int k1 = (t + 1) << 5;
            async_load_b128(asl[buf ^ 1] + loff0, Ar0 + k1 + c0);
            async_load_b128(asl[buf ^ 1] + loff1, Ar1 + k1 + c1);
            async_load_b128(bsl[buf ^ 1] + loff0, Br0 + k1 + c0);
            async_load_b128(bsl[buf ^ 1] + loff1, Br1 + k1 + c1);
            wait_async4();   // current tile's 4 done; next tile's 4 still flying
        } else {
            wait_async0();
        }
        __syncthreads();

        // fragments: lane half selects K subset {0..7,16..23} / {8..15,24..31}
        const int m  = lane & 15;
        const int ko = (lane < 16) ? 0 : 8;
        Frag af[2];
        #pragma unroll
        for (int i = 0; i < 2; ++i) {
            int row = wm * 32 + i * 16 + m;
            af[i].q[0] = *(const uint4*)&As[buf][row][ko];
            af[i].q[1] = *(const uint4*)&As[buf][row][ko + 16];
        }
        #pragma unroll
        for (int f = 0; f < 4; ++f) {
            Frag bfr;
            int n = wn * 64 + f * 16 + m;
            bfr.q[0] = *(const uint4*)&Bs[buf][n][ko];
            bfr.q[1] = *(const uint4*)&Bs[buf][n][ko + 16];
            #pragma unroll
            for (int i = 0; i < 2; ++i)
                acc[i][f] = __builtin_amdgcn_wmma_f32_16x16x32_bf16(
                    false, af[i].v, false, bfr.v, (short)0, acc[i][f], false, false);
        }
        __syncthreads();   // compute done; next iteration may overwrite this buffer
    }

    // C fragment: lane n = lane%16, rows r (+8 for upper lane half)
    #pragma unroll
    for (int i = 0; i < 2; ++i) {
        int mbase = bm + wm * 32 + i * 16 + ((lane < 16) ? 0 : 8);
        #pragma unroll
        for (int f = 0; f < 4; ++f) {
            int n = bn + wn * 64 + f * 16 + (lane & 15);
            #pragma unroll
            for (int r = 0; r < 8; ++r)
                C[(size_t)(mbase + r) * N + n] = acc[i][f][r];
        }
    }
}

// ---------------- RoPE on Q or K, scatter to [B, nheads, S, HD] bf16 ----------------
__global__ __launch_bounds__(256) void rope_qk(
    const float* __restrict__ qkv, unsigned short* __restrict__ dst,
    const int* __restrict__ positions, int nheads, int col_off)
{
    int idx = blockIdx.x * blockDim.x + threadIdx.x;
    int i = idx & 63;
    int h = (idx >> 6) % nheads;
    int s = (idx / (64 * nheads)) % SEQ;
    int b = idx / (64 * nheads * SEQ);
    const float* src = qkv + ((size_t)(b * SEQ + s)) * NQKV + col_off + h * HD;
    float x1 = src[i], x2 = src[i + 64];
    float pos = (float)positions[b * SEQ + s];
    float freq = __powf(10000.0f, -(float)(2 * i) * (1.0f / 128.0f));
    float sn, cs;
    __sincosf(pos * freq, &sn, &cs);
    unsigned short* d = dst + (((size_t)(b * nheads + h)) * SEQ + s) * HD;
    d[i]      = f2bf(x1 * cs - x2 * sn);
    d[i + 64] = f2bf(x2 * cs + x1 * sn);
}

// ---------------- V transpose to [B, NKV, HD, S] bf16 ----------------
__global__ __launch_bounds__(256) void vtransk(const float* __restrict__ qkv,
                                               unsigned short* __restrict__ Vt)
{
    int idx = blockIdx.x * blockDim.x + threadIdx.x;   // 22 bits total
    int d = idx & 127;
    int s = (idx >> 7) & 2047;
    int h = (idx >> 18) & 7;
    int b = idx >> 21;
    float v = qkv[((size_t)(b * SEQ + s)) * NQKV + QSZ + KVSZ + h * HD + d];
    Vt[(((size_t)(b * NKV + h)) * HD + d) * SEQ + s] = f2bf(v);
}

// ---------------- Flash attention: 1 wave = 16 queries, online softmax ----------------
// grid (B*NH, S/64), block 128 (4 waves). Qb/Kb: [B,*,S,HD]; Vt: [B,NKV,HD,S].
// K/V fragment loads come straight from global: each (b,kv-head) K/V slab is 512KB
// and reused by 128 workgroups, so it lives in MI455X's 192MB L2.
// Ob: [B*S, NH*HD] bf16 (row-major, ready to be GEMM-A for out-proj).
__global__ __launch_bounds__(128) void attnk(
    const unsigned short* __restrict__ Qb,
    const unsigned short* __restrict__ Kb,
    const unsigned short* __restrict__ Vt,
    unsigned short* __restrict__ Ob)
{
    __shared__ __align__(16) unsigned short pbuf[4][16 * 32];  // per-wave P transpose buffer

    const int tid  = threadIdx.x;
    const int lane = tid & 31;
    const int w    = tid >> 5;
    const int bh   = blockIdx.x;
    const int b    = bh >> 5;
    const int h    = bh & 31;
    const int hk   = h >> 2;                 // GQA group of 4
    const int q0   = blockIdx.y * 64 + w * 16;

    const unsigned short* Qp = Qb + (((size_t)(b * NH + h)) * SEQ + q0) * HD;
    const unsigned short* Kp = Kb + ((size_t)(b * NKV + hk)) * SEQ * HD;
    const unsigned short* Vp = Vt + ((size_t)(b * NKV + hk)) * HD * SEQ;

    const int m  = lane & 15;
    const int ko = (lane < 16) ? 0 : 8;

    // Q fragments: 16x128 as 4 chunks of 16x32
    Frag qf[4];
    #pragma unroll
    for (int c = 0; c < 4; ++c) {
        qf[c].q[0] = *(const uint4*)(Qp + (size_t)m * HD + c * 32 + ko);
        qf[c].q[1] = *(const uint4*)(Qp + (size_t)m * HD + c * 32 + ko + 16);
    }

    v8f acc[8] = {};                         // O: 16x128 fp32
    float rmax[8], rsum[8];
    #pragma unroll
    for (int r = 0; r < 8; ++r) { rmax[r] = -1e30f; rsum[r] = 0.0f; }
    const float scale = 0.08838834764831845f;    // 1/sqrt(128)
    const int qlim = q0 + 16;                    // causal: keys < qlim

    for (int kb = 0; kb < qlim; kb += 32) {
        // ---- scores: two 16x16 tiles over 32 keys ----
        v8f s0 = {}, s1 = {};
        #pragma unroll
        for (int c = 0; c < 4; ++c) {
            Frag kf0, kf1;
            const unsigned short* kr0 = Kp + (size_t)(kb + m) * HD + c * 32;
            const unsigned short* kr1 = Kp + (size_t)(kb + 16 + m) * HD + c * 32;
            kf0.q[0] = *(const uint4*)(kr0 + ko);
            kf0.q[1] = *(const uint4*)(kr0 + ko + 16);
            kf1.q[0] = *(const uint4*)(kr1 + ko);
            kf1.q[1] = *(const uint4*)(kr1 + ko + 16);
            s0 = __builtin_amdgcn_wmma_f32_16x16x32_bf16(false, qf[c].v, false, kf0.v, (short)0, s0, false, false);
            s1 = __builtin_amdgcn_wmma_f32_16x16x32_bf16(false, qf[c].v, false, kf1.v, (short)0, s1, false, false);
        }

        // ---- online softmax; row stats live in matching (lane-half, vgpr) slots ----
        const int n = lane & 15;
        const int mrow = (lane < 16) ? 0 : 8;
        #pragma unroll
        for (int r = 0; r < 8; ++r) {
            int qa = q0 + mrow + r;
            float v0 = s0[r] * scale; if (kb + n      > qa) v0 = -1e30f;
            float v1 = s1[r] * scale; if (kb + 16 + n > qa) v1 = -1e30f;
            float mx = fmaxf(v0, v1);
            #pragma unroll
            for (int off = 8; off; off >>= 1) mx = fmaxf(mx, __shfl_xor(mx, off, 16));
            float nm = fmaxf(rmax[r], mx);
            float corr = __expf(rmax[r] - nm);
            rmax[r] = nm;
            float p0 = __expf(v0 - nm), p1 = __expf(v1 - nm);
            float ps = p0 + p1;
            #pragma unroll
            for (int off = 8; off; off >>= 1) ps += __shfl_xor(ps, off, 16);
            rsum[r] = rsum[r] * corr + ps;
            #pragma unroll
            for (int f = 0; f < 8; ++f) acc[f][r] *= corr;   // rows align with O fragments
            int mm = mrow + r;
            pbuf[w][mm * 32 + n]      = f2bf(p0);            // P (16x32) row-major in LDS
            pbuf[w][mm * 32 + 16 + n] = f2bf(p1);
        }

        // ---- P (C-layout) -> A-layout via per-wave LDS; then O += P * V ----
        Frag pf;
        pf.q[0] = *(const uint4*)&pbuf[w][m * 32 + ko];
        pf.q[1] = *(const uint4*)&pbuf[w][m * 32 + ko + 16];
        #pragma unroll
        for (int f = 0; f < 8; ++f) {
            Frag vf;
            int d = f * 16 + (lane & 15);
            const unsigned short* vr = Vp + (size_t)d * SEQ;   // V^T row d, contiguous keys
            vf.q[0] = *(const uint4*)(vr + kb + ko);
            vf.q[1] = *(const uint4*)(vr + kb + 16 + ko);
            acc[f] = __builtin_amdgcn_wmma_f32_16x16x32_bf16(false, pf.v, false, vf.v, (short)0, acc[f], false, false);
        }
    }

    // ---- normalize and write O tile to [B*S, NH*HD] ----
    const int mbase = q0 + ((lane < 16) ? 0 : 8);
    const int nn = lane & 15;
    #pragma unroll
    for (int r = 0; r < 8; ++r) {
        float inv = 1.0f / rsum[r];
        size_t row = (size_t)b * SEQ + mbase + r;
        #pragma unroll
        for (int f = 0; f < 8; ++f)
            Ob[row * QSZ + h * HD + f * 16 + nn] = f2bf(acc[f][r] * inv);
    }
}

// ---------------- host-side orchestration ----------------
extern "C" void kernel_launch(void* const* d_in, const int* in_sizes, int n_in,
                              void* d_out, int out_size, void* d_ws, size_t ws_size,
                              hipStream_t stream) {
    (void)in_sizes; (void)n_in; (void)out_size; (void)ws_size;
    const int*   positions = (const int*)d_in[0];
    const float* hidden    = (const float*)d_in[1];
    const float* w_qkv     = (const float*)d_in[2];
    const float* w_o       = (const float*)d_in[3];
    float*       out       = (float*)d_out;

    size_t off = 0;
    auto alloc = [&](size_t bytes) -> void* {
        void* p = (char*)d_ws + off;
        off += (bytes + 255) & ~(size_t)255;
        return p;
    };
    const size_t MROWS = (size_t)BATCH * SEQ;                       // 4096
    unsigned short* Xb     = (unsigned short*)alloc(MROWS * HID * 2);
    unsigned short* Wqkvt  = (unsigned short*)alloc((size_t)NQKV * HID * 2);  // [N][K]
    unsigned short* Wot    = (unsigned short*)alloc((size_t)HID * QSZ * 2);   // [N][K]
    float*          qkvf   = (float*)alloc(MROWS * NQKV * 4);
    unsigned short* Qb     = (unsigned short*)alloc((size_t)BATCH * NH  * SEQ * HD * 2);
    unsigned short* Kb     = (unsigned short*)alloc((size_t)BATCH * NKV * SEQ * HD * 2);
    unsigned short* Vt     = (unsigned short*)alloc((size_t)BATCH * NKV * SEQ * HD * 2);
    unsigned short* Ab     = (unsigned short*)alloc(MROWS * QSZ * 2);

    // 1) cast activations; cast+transpose weights to [N][K] for B-fragment loads
    {
        int n4 = (int)(MROWS * HID / 4);
        castk<<<n4 / 256, 256, 0, stream>>>(hidden, Xb, n4);
        castT<<<dim3(NQKV / 32, HID / 32), 256, 0, stream>>>(w_qkv, Wqkvt, HID, NQKV);
        castT<<<dim3(HID / 32, QSZ / 32),  256, 0, stream>>>(w_o,   Wot,   QSZ, HID);
    }
    // 2) QKV projection: [4096,4096] x [4096,6144] -> f32
    gemm_bf16f32<<<dim3(NQKV / 128, (int)(MROWS / 128)), 256, 0, stream>>>(
        Xb, Wqkvt, qkvf, (int)MROWS, NQKV, HID);
    // 3) RoPE Q / K, V transpose
    rope_qk<<<(BATCH * SEQ * NH  * 64) / 256, 256, 0, stream>>>(qkvf, Qb, positions, NH, 0);
    rope_qk<<<(BATCH * SEQ * NKV * 64) / 256, 256, 0, stream>>>(qkvf, Kb, positions, NKV, QSZ);
    vtransk<<<(BATCH * NKV * SEQ * HD) / 256, 256, 0, stream>>>(qkvf, Vt);
    // 4) flash attention
    attnk<<<dim3(BATCH * NH, SEQ / 64), 128, 0, stream>>>(Qb, Kb, Vt, Ab);
    // 5) output projection: [4096,4096] x [4096,4096] -> f32 out
    gemm_bf16f32<<<dim3(HID / 128, (int)(MROWS / 128)), 256, 0, stream>>>(
        Ab, Wot, out, (int)MROWS, HID, QSZ);
}
